// NeoGNN_8083128451635
// MI455X (gfx1250) — matrix operand order repaired and checked by reading, not compiled, not run on hardware
//
#include <hip/hip_runtime.h>
#include <hip/hip_bf16.h>

// ---------------------------------------------------------------------------
// NeoGNN forward for MI455X (gfx1250, wave32, WMMA).
// Heavy path: 3x (8192x8192 adj) @ (8192x256) GEMMs in bf16 WMMA, fp32 acc;
// adj preconverted to bf16 (128 MiB -> L2-resident on 192 MiB L2).
// GEMM: 128x64 block tile, BK=32, double-buffered LDS, B pre-swizzled into
// WMMA fragment layout. sched_group_barrier pins the per-iteration pattern:
// 10x ds_load burst -> one dscnt drain -> 4x v_wmma back-to-back.
// ---------------------------------------------------------------------------

typedef __attribute__((ext_vector_type(16))) __bf16 v16bf;
typedef __attribute__((ext_vector_type(8)))  __bf16 v8bf;
typedef __attribute__((ext_vector_type(8)))  float  v8f;

static __device__ __forceinline__ unsigned short f2bf(float f) {
    unsigned int x = __builtin_bit_cast(unsigned int, f);
    unsigned int r = (x + 0x7FFFu + ((x >> 16) & 1u)) >> 16;   // RNE
    return (unsigned short)r;
}
static __device__ __forceinline__ float bf2f(unsigned short u) {
    unsigned int x = ((unsigned int)u) << 16;
    return __builtin_bit_cast(float, x);
}

// ------------------------- fp32 -> bf16 convert ----------------------------
__global__ __launch_bounds__(256)
void cvt_f32_bf16(const float* __restrict__ src, unsigned short* __restrict__ dst, long n) {
    long i = (long)blockIdx.x * 256 + threadIdx.x;
    if (i < n) dst[i] = f2bf(src[i]);
}

// ------------------------- bf16 WMMA GEMM ----------------------------------
// C[M,Ncols] = A[M,K](bf16) @ B[K,Ncols](bf16) (+bias) (relu?)
// 256 threads = 8 waves. Tile BM=128 x BN=64, BK=32.
// Wave w owns rows [w*16, w*16+16) and all 64 cols -> 4 accumulators.
#define BM 128
#define BN 64
#define BK 32
#define LDA_S 40   // A LDS row stride in bf16 (80 B, rows 16B-aligned)
#define BSTR 24    // B swizzled per-lane stride in bf16 (48 B, 16B-aligned)

static __device__ __forceinline__ v16bf ld_frag(const unsigned short* p, int gap) {
    v8bf lo = *(const v8bf*)p;
    v8bf hi = *(const v8bf*)(p + gap);
    return __builtin_shufflevector(lo, hi, 0,1,2,3,4,5,6,7,8,9,10,11,12,13,14,15);
}

template<bool BIAS, bool RELU, bool OUTF, bool OUTB>
__global__ __launch_bounds__(256)
void gemm_bf16_wmma(const unsigned short* __restrict__ A, int lda,
                    const unsigned short* __restrict__ B, int ldb,
                    const float* __restrict__ bias,
                    float* __restrict__ Cf, unsigned short* __restrict__ Cb,
                    int ldc, int K)
{
    __shared__ unsigned short As[2][BM * LDA_S];
    __shared__ unsigned short Bsw[2][4][32 * BSTR];

    const int tid  = threadIdx.x;
    const int wave = tid >> 5;
    const int lane = tid & 31;
    const int half = lane >> 4;
    const int l16  = lane & 15;

    const int row0 = blockIdx.y * BM;
    const int col0 = blockIdx.x * BN;

    // per-thread global-load slots
    const int ar = tid >> 1;               // A row 0..127
    const int ac = (tid & 1) * 16;         // A col 0 or 16
    const int br = tid >> 3;               // B row 0..31
    const int bc = (tid & 7) * 8;          // B col 0..56

    const unsigned short* Abase = A + (size_t)(row0 + ar) * lda + ac;
    const unsigned short* Bbase = B + (size_t)br * ldb + col0 + bc;

    v8f acc0 = {}, acc1 = {}, acc2 = {}, acc3 = {};

    uint4 a0, a1, b0;
    // prologue: stage tile 0
    a0 = *(const uint4*)(Abase);
    a1 = *(const uint4*)(Abase + 8);
    b0 = *(const uint4*)(Bbase);
    {
        *(uint4*)(&As[0][ar * LDA_S + ac])     = a0;
        *(uint4*)(&As[0][ar * LDA_S + ac + 8]) = a1;
        union { uint4 v; unsigned short s[8]; } bu; bu.v = b0;
        #pragma unroll
        for (int jj = 0; jj < 8; ++jj) {
            int c = bc + jj;
            Bsw[0][c >> 4][((br & 16) | (c & 15)) * BSTR + (br & 15)] = bu.s[jj];
        }
    }

    const int T = K / BK;
    for (int t = 0; t < T; ++t) {
        const int cur = t & 1, nxt = cur ^ 1;
        __syncthreads();
        // issue next tile's global loads early (overlap with WMMAs)
        if (t + 1 < T) {
            const size_t ko = (size_t)(t + 1) * BK;
            a0 = *(const uint4*)(Abase + ko);
            a1 = *(const uint4*)(Abase + ko + 8);
            b0 = *(const uint4*)(Bbase + ko * ldb);
        }

        // Burst-issue ALL fragment loads...
        // A fragment: lane m=wave*16+l16: af[0..15]=K half*16+j
        v16bf af  = ld_frag(&As[cur][(wave * 16 + l16) * LDA_S + half * 16], 8);
        // B fragments: pre-swizzled, per-lane contiguous (gap 8 elems = 16 B)
        v16bf bf0 = ld_frag(&Bsw[cur][0][lane * BSTR], 8);
        v16bf bf1 = ld_frag(&Bsw[cur][1][lane * BSTR], 8);
        v16bf bf2 = ld_frag(&Bsw[cur][2][lane * BSTR], 8);
        v16bf bf3 = ld_frag(&Bsw[cur][3][lane * BSTR], 8);

        acc0 = __builtin_amdgcn_wmma_f32_16x16x32_bf16(false, af, false, bf0,
                                                       (short)0, acc0, false, false);
        acc1 = __builtin_amdgcn_wmma_f32_16x16x32_bf16(false, af, false, bf1,
                                                       (short)0, acc1, false, false);
        acc2 = __builtin_amdgcn_wmma_f32_16x16x32_bf16(false, af, false, bf2,
                                                       (short)0, acc2, false, false);
        acc3 = __builtin_amdgcn_wmma_f32_16x16x32_bf16(false, af, false, bf3,
                                                       (short)0, acc3, false, false);

        // Pin the schedule: all 10 DS reads first (one dscnt drain), then the
        // 4 WMMAs back-to-back. Forces all B fragments live (no recycling).
        __builtin_amdgcn_sched_group_barrier(0x100, 10, 0);  // DS read x10
        __builtin_amdgcn_sched_group_barrier(0x008, 4, 0);   // WMMA x4

        // stage next tile into the other LDS buffer
        if (t + 1 < T) {
            *(uint4*)(&As[nxt][ar * LDA_S + ac])     = a0;
            *(uint4*)(&As[nxt][ar * LDA_S + ac + 8]) = a1;
            union { uint4 v; unsigned short s[8]; } bu; bu.v = b0;
            #pragma unroll
            for (int jj = 0; jj < 8; ++jj) {
                int c = bc + jj;
                Bsw[nxt][c >> 4][((br & 16) | (c & 15)) * BSTR + (br & 15)] = bu.s[jj];
            }
        }
    }

    // Epilogue: D layout — VGPR i -> row (i + 8*half), col = lane%16
    const int rbase = row0 + wave * 16 + half * 8;
    v8f accs[4] = {acc0, acc1, acc2, acc3};
    #pragma unroll
    for (int tl = 0; tl < 4; ++tl) {
        const int c = col0 + tl * 16 + l16;
        float bv = 0.0f;
        if (BIAS) bv = bias[c];
        #pragma unroll
        for (int i = 0; i < 8; ++i) {
            float v = accs[tl][i];
            if (BIAS) v += bv;
            if (RELU) v = v > 0.f ? v : 0.f;
            const size_t idx = (size_t)(rbase + i) * ldc + c;
            if (OUTF) Cf[idx] = v;
            if (OUTB) Cb[idx] = f2bf(v);
        }
    }
}

// --------------------- per-edge Hadamard h[row]*h[col] ---------------------
__global__ __launch_bounds__(256)
void edge_mul_kernel(const int* __restrict__ ei, const float* __restrict__ h,
                     unsigned short* __restrict__ hp, int E, int H) {
    int idx = blockIdx.x * 256 + threadIdx.x;
    if (idx >= E * H) return;
    int e = idx >> 8;            // H == 256
    int j = idx & 255;
    int r = ei[e], c = ei[E + e];
    hp[idx] = f2bf(h[(size_t)r * H + j] * h[(size_t)c * H + j]);
}

// --------------------- predictor last layer (H -> 1) -----------------------
__global__ __launch_bounds__(256)
void pred_last_kernel(const unsigned short* __restrict__ z2,
                      const float* __restrict__ W2, const float* __restrict__ b2,
                      float* __restrict__ out_feat, int E, int H) {
    int e    = blockIdx.x * 8 + (threadIdx.x >> 5);
    int lane = threadIdx.x & 31;
    if (e >= E) return;
    float s = 0.f;
    for (int j = lane; j < H; j += 32)
        s += bf2f(z2[(size_t)e * H + j]) * W2[j];
    #pragma unroll
    for (int off = 16; off > 0; off >>= 1) s += __shfl_down(s, off, 32);
    if (lane == 0) out_feat[e] = s + b2[0];
}

// --------------------- structural path helpers -----------------------------
__global__ __launch_bounds__(256)
void fe_scalar_kernel(const float* __restrict__ W0, const float* __restrict__ b0,
                      const float* __restrict__ W1, const float* __restrict__ b1,
                      float* __restrict__ c_out) {
    __shared__ float red[256];
    int j = threadIdx.x;
    float a = W0[j] + b0[j];             // input == 1.0
    a = a > 0.f ? a : 0.f;
    red[j] = a * W1[j];
    __syncthreads();
    for (int s = 128; s > 0; s >>= 1) {
        if (j < s) red[j] += red[j + s];
        __syncthreads();
    }
    if (j == 0) c_out[0] = red[0] + b1[0];
}

__global__ __launch_bounds__(256)
void zero_kernel(float* __restrict__ p, int n) {
    int i = blockIdx.x * 256 + threadIdx.x;
    if (i < n) p[i] = 0.f;
}

__global__ __launch_bounds__(256)
void indeg_kernel(const int* __restrict__ ei, float* __restrict__ ns, int E) {
    int e = blockIdx.x * 256 + threadIdx.x;
    if (e < E) atomicAdd(&ns[ei[E + e]], 1.0f);
}

// s_node[n] = MLP2(c * indeg[n]; fn_*) — scalar 1->256->1, weights in LDS
__global__ __launch_bounds__(256)
void node_mlp_kernel(const float* __restrict__ ns, const float* __restrict__ c_in,
                     const float* __restrict__ W0, const float* __restrict__ b0,
                     const float* __restrict__ W1, const float* __restrict__ b1,
                     float* __restrict__ s_node, int N) {
    __shared__ float w0[256], bb[256], w1[256];
    int t = threadIdx.x;
    w0[t] = W0[t]; bb[t] = b0[t]; w1[t] = W1[t];
    __syncthreads();
    int n = blockIdx.x * 256 + t;
    if (n >= N) return;
    float v = c_in[0] * ns[n];
    float s = b1[0];
    for (int j = 0; j < 256; ++j) {
        float a = v * w0[j] + bb[j];
        s += (a > 0.f ? a : 0.f) * w1[j];
    }
    s_node[n] = s;
}

// per-edge dot(adj[row], adj[col]) — one wave32 per edge, float4 streaming
__global__ __launch_bounds__(256)
void edge_dot_kernel(const int* __restrict__ ei, const float* __restrict__ adj,
                     float* __restrict__ dot, int E, int N) {
    int e    = blockIdx.x * 8 + (threadIdx.x >> 5);
    int lane = threadIdx.x & 31;
    if (e >= E) return;
    int r = ei[e], c = ei[E + e];
    const float4* pa = (const float4*)(adj + (size_t)r * N);
    const float4* pb = (const float4*)(adj + (size_t)c * N);
    float s = 0.f;
    int n4 = N >> 2;
    for (int i = lane; i < n4; i += 32) {
        float4 a = pa[i], b = pb[i];
        s += a.x * b.x + a.y * b.y + a.z * b.z + a.w * b.w;
    }
    #pragma unroll
    for (int off = 16; off > 0; off >>= 1) s += __shfl_down(s, off, 32);
    if (lane == 0) dot[e] = s;
}

// combine: gp MLP, sigmoid, softmax(alpha) blend
__global__ __launch_bounds__(256)
void final_kernel(const int* __restrict__ ei, const float* __restrict__ dot,
                  const float* __restrict__ s_node,
                  const float* __restrict__ gpW0, const float* __restrict__ gpb0,
                  const float* __restrict__ gpW1, const float* __restrict__ gpb1,
                  const float* __restrict__ alpha, const float* __restrict__ out_feat,
                  float* __restrict__ out, float* __restrict__ out_struct,
                  float* __restrict__ out_raw, int E) {
    __shared__ float w0[256], bb[256], w1[256];
    int t = threadIdx.x;
    w0[t] = gpW0[t]; bb[t] = gpb0[t]; w1[t] = gpW1[t];
    __syncthreads();
    int e = blockIdx.x * 256 + t;
    if (e >= E) return;
    int r = ei[e], c = ei[E + e];
    float v = dot[e] * s_node[r] * s_node[c];
    float raw = gpb1[0];
    for (int j = 0; j < 256; ++j) {
        float a = v * w0[j] + bb[j];
        raw += (a > 0.f ? a : 0.f) * w1[j];
    }
    float os = 1.f / (1.f + __expf(-raw));
    float a0 = alpha[0], a1 = alpha[1];
    float m  = fmaxf(a0, a1);
    float e0 = __expf(a0 - m), e1 = __expf(a1 - m);
    float inv = 1.f / (e0 + e1);
    float of  = out_feat[e];
    out[e]        = (e0 * inv) * os + (e1 * inv) * of + 1e-15f;
    out_struct[e] = os;
    out_raw[e]    = raw;
}

// ---------------------------------------------------------------------------
extern "C" void kernel_launch(void* const* d_in, const int* in_sizes, int n_in,
                              void* d_out, int out_size, void* d_ws, size_t ws_size,
                              hipStream_t stream) {
    (void)in_sizes; (void)n_in; (void)out_size; (void)ws_size;
    const int N = 8192, E = 16384, F = 128, H = 256;

    const int*   ei      = (const int*)d_in[0];
    const float* x       = (const float*)d_in[1];
    const float* adj     = (const float*)d_in[2];
    const float* gcn_W0  = (const float*)d_in[3];
    const float* gcn_b0  = (const float*)d_in[4];
    const float* gcn_W1  = (const float*)d_in[5];
    const float* gcn_b1  = (const float*)d_in[6];
    const float* gcn_W2  = (const float*)d_in[7];
    const float* gcn_b2  = (const float*)d_in[8];
    const float* pred_W0 = (const float*)d_in[9];
    const float* pred_b0 = (const float*)d_in[10];
    const float* pred_W1 = (const float*)d_in[11];
    const float* pred_b1 = (const float*)d_in[12];
    const float* pred_W2 = (const float*)d_in[13];
    const float* pred_b2 = (const float*)d_in[14];
    const float* fe_W0   = (const float*)d_in[15];
    const float* fe_b0   = (const float*)d_in[16];
    const float* fe_W1   = (const float*)d_in[17];
    const float* fe_b1   = (const float*)d_in[18];
    const float* fn_W0   = (const float*)d_in[19];
    const float* fn_b0   = (const float*)d_in[20];
    const float* fn_W1   = (const float*)d_in[21];
    const float* fn_b1   = (const float*)d_in[22];
    const float* gp_W0   = (const float*)d_in[23];
    const float* gp_b0   = (const float*)d_in[24];
    const float* gp_W1   = (const float*)d_in[25];
    const float* gp_b1   = (const float*)d_in[26];
    const float* alpha   = (const float*)d_in[27];

    float* out        = (float*)d_out;          // [E]
    float* out_struct = out + E;                // [E]
    float* out_feat   = out_struct + E;         // [E]
    float* out_raw    = out_feat + E;           // [E]

    // ------ workspace carve-out (256B aligned) ------
    size_t off = 0;
    auto alloc = [&](size_t bytes) -> void* {
        void* p = (char*)d_ws + off;
        off += (bytes + 255) & ~(size_t)255;
        return p;
    };
    unsigned short* adj_bf = (unsigned short*)alloc((size_t)N * N * 2);  // 128 MiB, L2-resident
    unsigned short* x_bf   = (unsigned short*)alloc((size_t)N * F * 2);
    unsigned short* w0_bf  = (unsigned short*)alloc((size_t)F * H * 2);
    unsigned short* w1_bf  = (unsigned short*)alloc((size_t)H * H * 2);
    unsigned short* w2_bf  = (unsigned short*)alloc((size_t)H * H * 2);
    unsigned short* pw0_bf = (unsigned short*)alloc((size_t)H * H * 2);
    unsigned short* pw1_bf = (unsigned short*)alloc((size_t)H * H * 2);
    unsigned short* t_bf   = (unsigned short*)alloc((size_t)N * H * 2);
    unsigned short* h_bf   = (unsigned short*)alloc((size_t)N * H * 2);
    float*          h_f32  = (float*)alloc((size_t)N * H * 4);
    unsigned short* hp_bf  = (unsigned short*)alloc((size_t)E * H * 2);  // reused as z2
    unsigned short* z1_bf  = (unsigned short*)alloc((size_t)E * H * 2);
    float*          c_sc   = (float*)alloc(4);
    float*          ns     = (float*)alloc((size_t)N * 4);
    float*          s_node = (float*)alloc((size_t)N * 4);
    float*          dot    = (float*)alloc((size_t)E * 4);

    // ------ 1) bf16 conversions ------
    auto cvt = [&](const float* s, unsigned short* d, long n) {
        cvt_f32_bf16<<<dim3((unsigned)((n + 255) / 256)), dim3(256), 0, stream>>>(s, d, n);
    };
    cvt(adj, adj_bf, (long)N * N);
    cvt(x, x_bf, (long)N * F);
    cvt(gcn_W0, w0_bf, (long)F * H);
    cvt(gcn_W1, w1_bf, (long)H * H);
    cvt(gcn_W2, w2_bf, (long)H * H);
    cvt(pred_W0, pw0_bf, (long)H * H);
    cvt(pred_W1, pw1_bf, (long)H * H);

    // ------ 2) GEMM launchers (compile-time specialized epilogues) ------
    auto gemm_plain = [&](const unsigned short* A, int lda, const unsigned short* B, int ldb,
                          unsigned short* Cb, int M, int Nc, int K) {
        gemm_bf16_wmma<false, false, false, true>
            <<<dim3(Nc / BN, M / BM), dim3(256), 0, stream>>>(
                A, lda, B, ldb, nullptr, nullptr, Cb, Nc, K);
    };
    auto gemm_br = [&](const unsigned short* A, int lda, const unsigned short* B, int ldb,
                       const float* bias, unsigned short* Cb, int M, int Nc, int K) {
        gemm_bf16_wmma<true, true, false, true>
            <<<dim3(Nc / BN, M / BM), dim3(256), 0, stream>>>(
                A, lda, B, ldb, bias, nullptr, Cb, Nc, K);
    };
    auto gemm_bf32 = [&](const unsigned short* A, int lda, const unsigned short* B, int ldb,
                         const float* bias, float* Cf, int M, int Nc, int K) {
        gemm_bf16_wmma<true, false, true, false>
            <<<dim3(Nc / BN, M / BM), dim3(256), 0, stream>>>(
                A, lda, B, ldb, bias, Cf, nullptr, Nc, K);
    };

    // ------ 3) GCN: h = adj @ (h @ W) + b, relu on first two layers ------
    gemm_plain(x_bf, F, w0_bf, H, t_bf, N, H, F);                 // t0 = x @ W0
    gemm_br(adj_bf, N, t_bf, H, gcn_b0, h_bf, N, H, N);           // h1 = relu(A t0 + b0)
    gemm_plain(h_bf, H, w1_bf, H, t_bf, N, H, H);                 // t1 = h1 @ W1
    gemm_br(adj_bf, N, t_bf, H, gcn_b1, h_bf, N, H, N);           // h2 = relu(A t1 + b1)
    gemm_plain(h_bf, H, w2_bf, H, t_bf, N, H, H);                 // t2 = h2 @ W2
    gemm_bf32(adj_bf, N, t_bf, H, gcn_b2, h_f32, N, H, N);        // h  = A t2 + b2 (fp32)

    // ------ 4) feature score: predictor(h[row] * h[col]) ------
    edge_mul_kernel<<<dim3((E * H) / 256), dim3(256), 0, stream>>>(ei, h_f32, hp_bf, E, H);
    gemm_br(hp_bf, H, pw0_bf, H, pred_b0, z1_bf, E, H, H);        // z1
    gemm_br(z1_bf, H, pw1_bf, H, pred_b1, hp_bf, E, H, H);        // z2 (reuse hp)
    pred_last_kernel<<<dim3(E / 8), dim3(256), 0, stream>>>(hp_bf, pred_W2, pred_b2,
                                                            out_feat, E, H);

    // ------ 5) structural score ------
    fe_scalar_kernel<<<dim3(1), dim3(256), 0, stream>>>(fe_W0, fe_b0, fe_W1, fe_b1, c_sc);
    zero_kernel<<<dim3(N / 256), dim3(256), 0, stream>>>(ns, N);
    indeg_kernel<<<dim3(E / 256), dim3(256), 0, stream>>>(ei, ns, E);
    node_mlp_kernel<<<dim3(N / 256), dim3(256), 0, stream>>>(ns, c_sc, fn_W0, fn_b0,
                                                             fn_W1, fn_b1, s_node, N);
    edge_dot_kernel<<<dim3(E / 8), dim3(256), 0, stream>>>(ei, adj, dot, E, N);
    final_kernel<<<dim3(E / 256), dim3(256), 0, stream>>>(ei, dot, s_node,
                                                          gp_W0, gp_b0, gp_W1, gp_b1,
                                                          alpha, out_feat,
                                                          out, out_struct, out_raw, E);
}
